// GraphVAE_53420803228324
// MI455X (gfx1250) — matrix operand order repaired and checked by reading, not compiled, not run on hardware
//
#include <hip/hip_runtime.h>

typedef __attribute__((ext_vector_type(16))) _Float16 v16h;
typedef __attribute__((ext_vector_type(8)))  _Float16 v8h;
typedef __attribute__((ext_vector_type(8)))  float    v8f;

#define NN   131072   // nodes
#define NE   2097152  // edges
#define HID  128
#define LAT  64

// ---------------- degree / normalization ----------------
__global__ void k_init_deg(float* __restrict__ deg) {
    int i = blockIdx.x * 256 + threadIdx.x;
    if (i < NN) deg[i] = 1.0f;                       // self loop
}

__global__ void k_deg_accum(const int* __restrict__ col, float* __restrict__ deg) {
    int e = blockIdx.x * 256 + threadIdx.x;
    if (e < NE) atomicAdd(&deg[col[e]], 1.0f);
}

__global__ void k_make_dinv(float* __restrict__ deg) {
    int i = blockIdx.x * 256 + threadIdx.x;
    if (i < NN) deg[i] = rsqrtf(deg[i]);             // deg buffer now holds dinv
}

// ---------------- weight pre-swizzle to WMMA B-fragment layout ----------------
// Fragment order: ((kt*8 + nt)*32 + lane)*16 + j
// lane L holds column n = nt*16 + (L&15); halves j=0..15 are K = kt*32 + (L>>4)*16 + j
__global__ void k_swizzleW(const float* __restrict__ W, _Float16* __restrict__ out, int K) {
    int idx = blockIdx.x * 256 + threadIdx.x;
    int total = (K / 32) * 8 * 32 * 16;
    if (idx >= total) return;
    int j  = idx & 15;
    int L  = (idx >> 4) & 31;
    int t  = idx >> 9;
    int nt = t & 7;
    int kt = t >> 3;
    int k  = kt * 32 + (L >> 4) * 16 + j;
    int n  = nt * 16 + (L & 15);
    out[idx] = (_Float16)W[k * HID + n];
}

// ---------------- layer 1 projection: hw1 = x @ W1  (K = 3) ----------------
__global__ void k_gemm1(const float* __restrict__ x, const float* __restrict__ W1,
                        float* __restrict__ hw) {
    int idx = blockIdx.x * 256 + threadIdx.x;        // NN*HID threads
    int n = idx >> 7, c = idx & 127;
    float x0 = x[n * 3 + 0], x1 = x[n * 3 + 1], x2 = x[n * 3 + 2];
    hw[idx] = x0 * W1[c] + x1 * W1[HID + c] + x2 * W1[2 * HID + c];
}

// ---------------- edge scatter: agg[col] += hw[row] * dinv[row]*dinv[col] ----------------
__global__ void k_scatter(const float* __restrict__ hw, const float* __restrict__ dinv,
                          const int* __restrict__ row, const int* __restrict__ col,
                          float* __restrict__ agg) {
    long long t = (long long)blockIdx.x * 256 + threadIdx.x;
    int e    = (int)(t >> 5);
    int lane = (int)(t & 31);
    if (e >= NE) return;
    int r = row[e], c = col[e];
    float en = dinv[r] * dinv[c];
    const float4 v = *(const float4*)(hw + (size_t)r * HID + lane * 4);
    float* dst = agg + (size_t)c * HID + lane * 4;
    atomicAdd(dst + 0, v.x * en);
    atomicAdd(dst + 1, v.y * en);
    atomicAdd(dst + 2, v.z * en);
    atomicAdd(dst + 3, v.w * en);
}

// ---------------- finish layer 1: h1 = relu(agg + hw*dinv^2 + b1), store f16 ----------------
__global__ void k_finish1(const float* __restrict__ agg, const float* __restrict__ hw,
                          const float* __restrict__ dinv, const float* __restrict__ b1,
                          _Float16* __restrict__ h1) {
    int idx = blockIdx.x * 256 + threadIdx.x;        // NN*HID
    int n = idx >> 7, c = idx & 127;
    float dn = dinv[n];
    float v = agg[idx] + hw[idx] * dn * dn + b1[c];
    h1[idx] = (_Float16)fmaxf(v, 0.0f);
}

// ---------------- WMMA GEMM: out[M=NN, 128] = A[M,K](f16) @ Bsw(f16) (+bias, relu) ----------
template <int K, bool RELU, bool OUTF32, bool OUTF16>
__global__ __launch_bounds__(256) void k_gemm_wmma(const _Float16* __restrict__ A,
                                                   const _Float16* __restrict__ Bsw,
                                                   const float* __restrict__ bias,
                                                   float* __restrict__ outf,
                                                   _Float16* __restrict__ outh) {
    constexpr int KT = K / 32;
    int wave = threadIdx.x >> 5;
    int lane = threadIdx.x & 31;
    int rowbase = (blockIdx.x * 8 + wave) * 16;
    int mrow = rowbase + (lane & 15);
    int hi = lane >> 4;                              // lane group (upper half of wave)

    // A fragments in contiguous K order (matches swizzled B):
    //   hi=0 lanes: K [0..15];  hi=1 lanes: K [16..31]
    v16h afr[KT];
#pragma unroll
    for (int kt = 0; kt < KT; ++kt) {
        const _Float16* p = A + (size_t)mrow * K + kt * 32 + hi * 16;
        v8h c0 = *(const v8h*)(p);
        v8h c1 = *(const v8h*)(p + 8);
#pragma unroll
        for (int j = 0; j < 8; ++j) { afr[kt][j] = c0[j]; afr[kt][8 + j] = c1[j]; }
    }

#pragma unroll
    for (int nt = 0; nt < 8; ++nt) {                 // 8 column tiles of 16 -> 128 cols
        v8f acc = {};
#pragma unroll
        for (int kt = 0; kt < KT; ++kt) {
            const _Float16* bp = Bsw + ((size_t)((kt * 8 + nt) * 32 + lane) << 4);
            v8h b0 = *(const v8h*)(bp);
            v8h b1v = *(const v8h*)(bp + 8);
            v16h bfr;
#pragma unroll
            for (int j = 0; j < 8; ++j) { bfr[j] = b0[j]; bfr[8 + j] = b1v[j]; }
            acc = __builtin_amdgcn_wmma_f32_16x16x32_f16(
                false, afr[kt], false, bfr, (short)0, acc, false, false);
        }
        int colb = nt * 16 + (lane & 15);
        float bv = bias ? bias[colb] : 0.0f;
#pragma unroll
        for (int r = 0; r < 8; ++r) {                // D: VGPR r -> row r (lo) / 8+r (hi)
            float val = acc[r] + bv;
            if (RELU) val = fmaxf(val, 0.0f);
            int rr = rowbase + r + hi * 8;
            size_t o = (size_t)rr * HID + colb;
            if (OUTF32) outf[o] = val;
            if (OUTF16) outh[o] = (_Float16)val;
        }
    }
}

// ---------------- finish layer 2: mean/logvar out, z = mean + eps*exp(0.5*logvar) ----------
__global__ void k_finish2(const float* __restrict__ agg, const float* __restrict__ hw2,
                          const float* __restrict__ dinv, const float* __restrict__ b2,
                          const float* __restrict__ eps, float* __restrict__ mean_out,
                          float* __restrict__ logv_out, _Float16* __restrict__ zh) {
    int idx = blockIdx.x * 256 + threadIdx.x;        // NN*LAT
    int n = idx >> 6, c = idx & 63;
    float dn = dinv[n], sn = dn * dn;
    size_t base = (size_t)n * HID;
    float m  = agg[base + c]       + hw2[base + c]       * sn + b2[c];
    float lv = agg[base + 64 + c]  + hw2[base + 64 + c]  * sn + b2[64 + c];
    mean_out[idx] = m;
    logv_out[idx] = lv;
    zh[idx] = (_Float16)(m + eps[idx] * __expf(0.5f * lv));
}

// ---------------- recon = hd @ Wd2 + bd2  (3 output cols -> VALU) ----------------
__global__ void k_recon(const _Float16* __restrict__ hd, const float* __restrict__ Wd2,
                        const float* __restrict__ bd2, float* __restrict__ out) {
    int idx = blockIdx.x * 256 + threadIdx.x;        // NN*3
    if (idx >= NN * 3) return;
    int n = idx / 3, d = idx % 3;
    const _Float16* h = hd + (size_t)n * HID;
    float s = bd2[d];
#pragma unroll 8
    for (int k = 0; k < HID; ++k) s += (float)h[k] * Wd2[k * 3 + d];
    out[idx] = s;
}

extern "C" void kernel_launch(void* const* d_in, const int* in_sizes, int n_in,
                              void* d_out, int out_size, void* d_ws, size_t ws_size,
                              hipStream_t stream) {
    (void)in_sizes; (void)n_in; (void)out_size; (void)ws_size;
    const float* x    = (const float*)d_in[0];
    const float* eps  = (const float*)d_in[1];
    const float* W1   = (const float*)d_in[2];
    const float* b1   = (const float*)d_in[3];
    const float* W2   = (const float*)d_in[4];
    const float* b2   = (const float*)d_in[5];
    const float* Wd1  = (const float*)d_in[6];
    const float* bd1  = (const float*)d_in[7];
    const float* Wd2  = (const float*)d_in[8];
    const float* bd2  = (const float*)d_in[9];
    const int*   ei   = (const int*)d_in[10];
    const int*   erow = ei;            // edge_index[0]
    const int*   ecol = ei + NE;       // edge_index[1]

    // workspace layout (bytes)
    char* ws = (char*)d_ws;
    float*     deg   = (float*)(ws + 0x0000000);            // NN*4            (-> dinv)
    _Float16*  W2sw  = (_Float16*)(ws + 0x0080000);         // 128*128*2
    _Float16*  Wd1sw = (_Float16*)(ws + 0x0088000);         // 64*128*2
    float*     hw    = (float*)(ws + 0x0090000);            // NN*128*4 = 64MB (hw1 then hw2)
    float*     agg   = (float*)(ws + 0x4090000);            // NN*128*4 = 64MB
    _Float16*  h1h   = (_Float16*)(ws + 0x8090000);         // NN*128*2 = 32MB (h1 then z)
    _Float16*  hdh   = (_Float16*)agg;                      // reuse agg after finish2

    float* out   = (float*)d_out;
    float* recon = out;                        // NN*3
    float* meanO = out + (size_t)NN * 3;       // NN*64
    float* logvO = meanO + (size_t)NN * LAT;   // NN*64

    const size_t aggBytes = (size_t)NN * HID * sizeof(float);

    // normalization
    k_init_deg <<<NN / 256, 256, 0, stream>>>(deg);
    k_deg_accum<<<NE / 256, 256, 0, stream>>>(ecol, deg);
    k_make_dinv<<<NN / 256, 256, 0, stream>>>(deg);

    // weight pre-swizzle (tiny)
    k_swizzleW<<<64, 256, 0, stream>>>(W2,  W2sw,  128);
    k_swizzleW<<<32, 256, 0, stream>>>(Wd1, Wd1sw, 64);

    // layer 1
    k_gemm1<<<(NN * HID) / 256, 256, 0, stream>>>(x, W1, hw);
    hipMemsetAsync(agg, 0, aggBytes, stream);
    k_scatter<<<(int)(((long long)NE * 32) / 256), 256, 0, stream>>>(hw, deg, erow, ecol, agg);
    k_finish1<<<(NN * HID) / 256, 256, 0, stream>>>(agg, hw, deg, b1, h1h);

    // layer 2: hw2 = h1 @ W2 via WMMA (no bias / no relu here)
    k_gemm_wmma<128, false, true, false><<<NN / 128, 256, 0, stream>>>(h1h, W2sw, nullptr, hw, nullptr);
    hipMemsetAsync(agg, 0, aggBytes, stream);
    k_scatter<<<(int)(((long long)NE * 32) / 256), 256, 0, stream>>>(hw, deg, erow, ecol, agg);
    k_finish2<<<(NN * LAT) / 256, 256, 0, stream>>>(agg, hw, deg, b2, eps, meanO, logvO, h1h);

    // decoder: hd = relu(z @ Wd1 + bd1) via WMMA (K=64), then recon
    k_gemm_wmma<64, true, false, true><<<NN / 128, 256, 0, stream>>>(h1h, Wd1sw, bd1, nullptr, hdh);
    k_recon<<<(NN * 3 + 255) / 256, 256, 0, stream>>>(hdh, Wd2, bd2, recon);
}
